// LigerLMHeadORPO_5583457485413
// MI455X (gfx1250) — compile-verified, matrix operand order checked by reading, not run on hardware
//
#include <hip/hip_runtime.h>
#include <stdint.h>

// ---------------------------------------------------------------------------
// Fused LM-head + ORPO loss for MI455X (gfx1250, wave32, WMMA + async-LDS).
//   logits GEMM: M=4096 tokens, N=32000 vocab, K=2048 -> 537 GFLOP, never
//   materialized. Pass 0 converts x/W to bf16 once. Pass 1: double-buffered
//   WMMA GEMM, tiles fetched with global_load_async_to_lds_b128 (ASYNCcnt),
//   compile-time buffer indices (K-loop unrolled by 2), one-ahead-pipelined
//   B fragments (dscnt overlap), ds_swizzle butterfly reductions.
// ---------------------------------------------------------------------------

typedef __attribute__((ext_vector_type(16))) __bf16       v16bf;
typedef __attribute__((ext_vector_type(8)))  float        v8f;
typedef __attribute__((ext_vector_type(4)))  unsigned int u32x4;
typedef __attribute__((ext_vector_type(4)))  float        f32x4;

#define H_DIM   2048
#define V_DIM   32000
#define M_TOK   4096            // 8 * 512
#define SEQ_T   512
#define BM      64
#define BN      256
#define BK      32
#define KSTEPS  (H_DIM / BK)    // 64
#define NTILES  (V_DIM / BN)    // 125
#define NPART   (2 * NTILES)    // 250 partials per token
#define PSTRIDE 256
#define LDSPAD  40              // bf16/row stride: 80B -> conflict-free b128

__device__ __forceinline__ unsigned int f2bf1(float f) {
  unsigned int u = __float_as_uint(f);
  u += 0x7FFFu + ((u >> 16) & 1u);        // round-to-nearest-even
  return u >> 16;
}
__device__ __forceinline__ unsigned int packbf(float lo, float hi) {
  return f2bf1(lo) | (f2bf1(hi) << 16);
}
// LDS byte address (generic LDS-aperture address truncates to LDS offset)
__device__ __forceinline__ unsigned ldsAddr(const void* p) {
  return (unsigned)(uintptr_t)p;
}
// CDNA5 async copy: 16B global -> LDS, tracked by ASYNCcnt (no VGPR data)
__device__ __forceinline__ void asyncLoad16(unsigned lds, unsigned long long g) {
  asm volatile("global_load_async_to_lds_b128 %0, %1, off"
               :: "v"(lds), "v"(g) : "memory");
}
__device__ __forceinline__ void waitAsync0() {
  asm volatile("s_wait_asynccnt 0" ::: "memory");
}
// butterfly shuffle via DS_SWIZZLE_B32 (group-of-32 xor mode, one DS op)
template <int MASK>
__device__ __forceinline__ float swzf(float v) {
  return __int_as_float(
      __builtin_amdgcn_ds_swizzle(__float_as_int(v), (MASK << 10) | 0x1F));
}
__device__ __forceinline__ float redmax16(float m) {   // within 16-lane half
  m = fmaxf(m, swzf<1>(m)); m = fmaxf(m, swzf<2>(m));
  m = fmaxf(m, swzf<4>(m)); m = fmaxf(m, swzf<8>(m));
  return m;
}
__device__ __forceinline__ float redsum16(float s) {
  s += swzf<1>(s); s += swzf<2>(s); s += swzf<4>(s); s += swzf<8>(s);
  return s;
}
__device__ __forceinline__ float redmax32(float m) {
  m = fmaxf(m, swzf<16>(m)); return redmax16(m);
}
__device__ __forceinline__ float redsum32(float s) {
  s += swzf<16>(s); return redsum16(s);
}

union Frag { u32x4 q[2]; v16bf v; };

// ---------------------------------------------------------------------------
// Pass 0: fp32 -> bf16 (RNE), 8 elements (2x f32x4 -> 1x u32x4) per thread.
// ---------------------------------------------------------------------------
__global__ __launch_bounds__(256)
void cvt_bf16_kernel(const float* __restrict__ src, u32x4* __restrict__ dst) {
  const size_t i = (size_t)blockIdx.x * blockDim.x + threadIdx.x;
  const f32x4* s = (const f32x4*)src;
  const f32x4 a = s[2 * i];
  const f32x4 b = s[2 * i + 1];
  u32x4 o;
  o[0] = packbf(a[0], a[1]);
  o[1] = packbf(a[2], a[3]);
  o[2] = packbf(b[0], b[1]);
  o[3] = packbf(b[2], b[3]);
  dst[i] = o;
}

// ---------------------------------------------------------------------------
// Pass 1: double-buffered WMMA GEMM tile + online-softmax partials.
// grid = (125, 64); block = 256 = 8 waves (4 along M x 2 along N).
// ---------------------------------------------------------------------------
__global__ __launch_bounds__(256)
void orpo_logit_tile_kernel(const unsigned short* __restrict__ xb,
                            const long long* __restrict__ y,
                            const unsigned short* __restrict__ Wb,
                            float* __restrict__ pmax,
                            float* __restrict__ psum,
                            float* __restrict__ lab) {
  __shared__ __attribute__((aligned(16))) unsigned short As[2][BM * LDSPAD];
  __shared__ __attribute__((aligned(16))) unsigned short Bs[2][BN * LDSPAD];

  const int tid   = threadIdx.x;
  const int lane  = tid & 31;
  const int wave  = tid >> 5;
  const int waveM = wave & 3;
  const int waveN = wave >> 2;
  const int mBase = blockIdx.y * BM;
  const int nBase = blockIdx.x * BN;
  const int col16 = lane & 15;
  const int half  = lane >> 4;

  v8f acc[8];
  #pragma unroll
  for (int t = 0; t < 8; ++t) { v8f z = {}; acc[t] = z; }

  // async staging: A = one 16B chunk/thread, B = one 64B row (4 chunks)/thread
  const int ar = tid >> 2;
  const int ac = (tid & 3) * 8;
  const unsigned long long aG =
      (unsigned long long)(uintptr_t)(xb + (size_t)(mBase + ar) * H_DIM + ac);
  const unsigned long long bG =
      (unsigned long long)(uintptr_t)(Wb + (size_t)(nBase + tid) * H_DIM);
  const unsigned aL[2] = { ldsAddr(&As[0][ar * LDSPAD + ac]),
                           ldsAddr(&As[1][ar * LDSPAD + ac]) };
  const unsigned bL[2] = { ldsAddr(&Bs[0][tid * LDSPAD]),
                           ldsAddr(&Bs[1][tid * LDSPAD]) };

  // LDS fragment offsets (16-bit A/B layout: lane = {M|N}%16, K split by half)
  const int aOff     = (waveM * 16 + col16) * LDSPAD + 8 * half;
  const int bOffBase = (waveN * 128 + col16) * LDSPAD + 8 * half;

  // one K-step on buffer CBUF; prefetch next tile into buffer NBUF
  auto kstep = [&](int it, auto cbufc, auto nbufc) {
    constexpr int CBUF = decltype(cbufc)::value;
    constexpr int NBUF = decltype(nbufc)::value;
    if (it + 1 < KSTEPS) {
      const unsigned long long ko = 2ull * (unsigned long long)((it + 1) * BK);
      asyncLoad16(aL[NBUF], aG + ko);
      #pragma unroll
      for (int c = 0; c < 4; ++c)
        asyncLoad16(bL[NBUF] + 16 * c, bG + ko + 16 * c);
    }
    const unsigned short* Ab = As[CBUF];
    const unsigned short* Bb = Bs[CBUF];
    Frag a;
    a.q[0] = *(const u32x4*)&Ab[aOff];
    a.q[1] = *(const u32x4*)&Ab[aOff + 16];
    Frag bfr[2];
    bfr[0].q[0] = *(const u32x4*)&Bb[bOffBase];
    bfr[0].q[1] = *(const u32x4*)&Bb[bOffBase + 16];
    #pragma unroll
    for (int nt = 0; nt < 8; ++nt) {
      if (nt < 7) {                          // one-ahead B-fragment pipeline
        const int bo = bOffBase + (nt + 1) * 16 * LDSPAD;
        bfr[(nt + 1) & 1].q[0] = *(const u32x4*)&Bb[bo];
        bfr[(nt + 1) & 1].q[1] = *(const u32x4*)&Bb[bo + 16];
      }
      acc[nt] = __builtin_amdgcn_wmma_f32_16x16x32_bf16(
          false, a.v, false, bfr[nt & 1].v, (short)0, acc[nt], false, false);
    }
    waitAsync0();                            // own copies for NBUF done
    __syncthreads();                         // everyone's copies + LDS reads
  };

  // prologue: tile 0 -> buffer 0
  asyncLoad16(aL[0], aG);
  #pragma unroll
  for (int c = 0; c < 4; ++c) asyncLoad16(bL[0] + 16 * c, bG + 16 * c);
  waitAsync0();
  __syncthreads();

  for (int it = 0; it < KSTEPS; it += 2) {   // compile-time buffer indices
    kstep(it,     std::integral_constant<int, 0>{}, std::integral_constant<int, 1>{});
    kstep(it + 1, std::integral_constant<int, 1>{}, std::integral_constant<int, 0>{});
  }

  // ---- online-softmax partials over this wave's 128-column slice ----
  // C layout: lanes 0-15 rows v, lanes 16-31 rows v+8; col = lane%16 + 16*t.
  #pragma unroll
  for (int v = 0; v < 8; ++v) {
    float m = -__builtin_inff();
    #pragma unroll
    for (int t = 0; t < 8; ++t) m = fmaxf(m, acc[t][v]);
    m = redmax16(m);                         // reduce within 16-lane half
    float s = 0.f;
    #pragma unroll
    for (int t = 0; t < 8; ++t) s += __expf(acc[t][v] - m);
    s = redsum16(s);

    const int mg = mBase + waveM * 16 + half * 8 + v;
    if (col16 == 0) {
      const int p = blockIdx.x * 2 + waveN;
      pmax[(size_t)mg * PSTRIDE + p] = m;
      psum[(size_t)mg * PSTRIDE + p] = s;
    }
    // label logit: exactly one lane in one block owns (row mg, col label)
    const long long lbl = y[mg];
    const int c = (int)lbl - (nBase + waveN * 128);
    if (lbl >= 0 && c >= 0 && c < 128 && (c & 15) == col16) {
      const int t = c >> 4;
      float lv = 0.f;
      #pragma unroll
      for (int tt = 0; tt < 8; ++tt) if (tt == t) lv = acc[tt][v];
      lab[mg] = lv;
    }
  }
}

// ---------------------------------------------------------------------------
// Pass 2: merge 250 partials/token -> per_tok = logit[label] - logsumexp.
// ---------------------------------------------------------------------------
__global__ __launch_bounds__(256)
void orpo_token_reduce_kernel(const long long* __restrict__ y,
                              const float* __restrict__ pmax,
                              const float* __restrict__ psum,
                              const float* __restrict__ lab,
                              float* __restrict__ ptok,
                              float* __restrict__ pmsk) {
  const int lane  = threadIdx.x & 31;
  const int token = blockIdx.x * 8 + (threadIdx.x >> 5);
  const float* pm = pmax + (size_t)token * PSTRIDE;
  const float* ps = psum + (size_t)token * PSTRIDE;

  float m = -__builtin_inff();
  for (int i = lane; i < NPART; i += 32) m = fmaxf(m, pm[i]);
  m = redmax32(m);
  float s = 0.f;
  for (int i = lane; i < NPART; i += 32) s += ps[i] * __expf(pm[i] - m);
  s = redsum32(s);

  if (lane == 0) {
    const long long lbl = y[token];
    const bool msk = (lbl != -100LL);
    ptok[token] = msk ? (lab[token] - (m + __logf(s))) : 0.f;
    pmsk[token] = msk ? 1.f : 0.f;
  }
}

// ---------------------------------------------------------------------------
// Pass 3: per-sequence averages + NLL + ORPO preference term -> scalar loss.
// ---------------------------------------------------------------------------
__global__ __launch_bounds__(256)
void orpo_final_kernel(const float* __restrict__ ptok,
                       const float* __restrict__ pmsk,
                       float* __restrict__ out) {
  __shared__ float sp[8], sm[8];
  const int lane = threadIdx.x & 31;
  const int b    = threadIdx.x >> 5;

  float s = 0.f, c = 0.f;
  for (int i = lane; i < SEQ_T; i += 32) {
    s += ptok[b * SEQ_T + i];
    c += pmsk[b * SEQ_T + i];
  }
  s = redsum32(s);
  c = redsum32(c);
  if (lane == 0) { sp[b] = s; sm[b] = c; }
  __syncthreads();

  if (threadIdx.x == 0) {
    float avg[8];
    float nnum = 0.f, nden = 0.f;
    for (int i = 0; i < 8; ++i) avg[i] = sp[i] / sm[i];
    for (int i = 0; i < 4; ++i) { nnum += sp[i]; nden += sm[i]; }
    const float nll = -nnum / nden;
    float acc = 0.f;
    for (int i = 0; i < 4; ++i) {
      const float ch = avg[i], rj = avg[4 + i];
      const float lo = ch - rj - (log1pf(-expf(ch)) - log1pf(-expf(rj)));
      acc += -log1pf(expf(-lo));             // log_sigmoid(lo)
    }
    out[0] = nll + (-0.1f) * acc / 4.f;
  }
}

// ---------------------------------------------------------------------------
extern "C" void kernel_launch(void* const* d_in, const int* in_sizes, int n_in,
                              void* d_out, int out_size, void* d_ws, size_t ws_size,
                              hipStream_t stream) {
  (void)in_sizes; (void)n_in; (void)out_size; (void)ws_size;
  const float*     x = (const float*)d_in[0];
  const long long* y = (const long long*)d_in[1];   // int64 labels
  const float*     W = (const float*)d_in[2];

  // workspace layout (~156.3 MB): bf16 copies of x and W, then fp32 partials
  unsigned short* xb = (unsigned short*)d_ws;
  unsigned short* Wb = xb + (size_t)M_TOK * H_DIM;
  float* fws  = (float*)(Wb + (size_t)V_DIM * H_DIM);
  float* pmax = fws;
  float* psum = pmax + (size_t)M_TOK * PSTRIDE;
  float* lab  = psum + (size_t)M_TOK * PSTRIDE;
  float* ptok = lab  + M_TOK;
  float* pmsk = ptok + M_TOK;

  // Pass 0: one-time bf16 conversion (8 floats/thread, exact division)
  cvt_bf16_kernel<<<((size_t)M_TOK * H_DIM) / 2048, 256, 0, stream>>>(
      x, (u32x4*)xb);
  cvt_bf16_kernel<<<((size_t)V_DIM * H_DIM) / 2048, 256, 0, stream>>>(
      W, (u32x4*)Wb);

  // Pass 1: fused GEMM + online softmax partials
  dim3 g1(NTILES, M_TOK / BM);                      // (125, 64)
  orpo_logit_tile_kernel<<<g1, 256, 0, stream>>>(xb, y, Wb, pmax, psum, lab);

  // Pass 2 + 3: token logsumexp merge, then scalar ORPO loss
  orpo_token_reduce_kernel<<<M_TOK / 8, 256, 0, stream>>>(y, pmax, psum, lab,
                                                          ptok, pmsk);
  orpo_final_kernel<<<1, 256, 0, stream>>>(ptok, pmsk, (float*)d_out);
}
